// HeteroRGCN_35038343201327
// MI455X (gfx1250) — matrix operand order repaired and checked by reading, not compiled, not run on hardware
//
#include <hip/hip_runtime.h>
#include <hip/hip_bf16.h>

// ---------------------------------------------------------------------------
// HeteroRGCN on MI455X (gfx1250, wave32, WMMA)
//   - all x128 GEMMs via v_wmma_f32_16x16x32_bf16 (bf16 in, f32 accum)
//   - LDS tiles stored in WMMA *fragment layout*: fragment build is
//     2x ds_load_b128 per fragment; all B frags hoisted so WMMAs issue
//     back-to-back against partial s_wait_dscnt
//   - staging: clamped rows (no EXEC masking) + uniform K-tail split
//   - mean aggregation via fp32 global atomics (dest buffers are L2-resident)
// ---------------------------------------------------------------------------

#define NU 50000
#define NP 100000
#define NS 500
#define HID 128
#define E_UP 500000
#define E_BT 100000

typedef __attribute__((ext_vector_type(16))) __bf16          v16bf;
typedef __attribute__((ext_vector_type(16))) unsigned short  v16u;
typedef __attribute__((ext_vector_type(8)))  float           v8f;

union F16frag { uint4 q[2]; v16u u; };

__device__ __forceinline__ unsigned short f2bf(float f) {
    union { float f; unsigned int u; } c; c.f = f;
    unsigned int u = c.u;
    unsigned int r = u + 0x7FFFu + ((u >> 16) & 1u);   // round-to-nearest-even
    return (unsigned short)(r >> 16);
}

__device__ __forceinline__ unsigned int pack2bf(float a, float b) {
    return (unsigned int)f2bf(a) | ((unsigned int)f2bf(b) << 16);
}

// ---------------------------------------------------------------------------
// Generic GEMM: C[M,128] = A[M,K] @ W[K,128] + bias  (opt: += C, opt: ReLU)
// Block = 256 threads = 8 waves. Block tile: 128 rows x 128 cols.
// Each wave owns a 16-row stripe and all 8 column tiles (8 x v8f accums).
//
// LDS holds tiles pre-swizzled into per-lane WMMA fragments (contiguous K, A and B alike):
//  A 16x32 bf16 frag, lane l: row = l&15; half=l>>4 selects
//     elems 0-15 = K (half*16..half*16+15), same order as B.
//  B 32x16 bf16 frag, lane l: col = l&15; kbase=(l>>4)*16; elem e = K kbase+e.
// ---------------------------------------------------------------------------
template<int ACC, int RELU>
__global__ __launch_bounds__(256) void gemm_bf16_wmma(
    const float* __restrict__ A, int K,
    const float* __restrict__ W, const float* __restrict__ bias,
    float* __restrict__ C, int M)
{
    // A fragments: [row][half] -> 16 contiguous ushort, row stride 40 (16B-aligned,
    // 20-bank stride => perfect 2-way bank pattern for ds_load_b128 across wave32)
    __shared__ unsigned short AsF[128 * 40];
    // B fragments: [ctile][lane] -> 16 contiguous ushort (fully dense, 16B-aligned)
    __shared__ unsigned short BsF[8 * 32 * 16];

    const int tid  = threadIdx.x;
    const int wave = tid >> 5;
    const int lane = tid & 31;
    const int half = lane >> 4;
    const int l15  = lane & 15;
    const int rowB = blockIdx.x * 128;
    const int wrow = wave * 16;
    const int rMax = M - 1;                // row clamp: OOB rows never stored

    v8f acc[8] = {};

    for (int kk = 0; kk < K; kk += 32) {
        if (kk + 32 <= K) {
            // ---- full-chunk fast path: unconditional, vector A loads ----
            #pragma unroll
            for (int idx = tid; idx < 128 * 16; idx += 256) {  // (row, kpair)
                int r = idx >> 4, k = (idx & 15) * 2;
                int gr = rowB + r; gr = gr > rMax ? rMax : gr;
                float2 v = *(const float2*)&A[(long)gr * K + kk + k];  // b64
                int h = k >> 4;
                int e = k & 15;                                // even
                *(unsigned int*)&AsF[r * 40 + h * 16 + e] = pack2bf(v.x, v.y);
            }
            #pragma unroll
            for (int idx = tid; idx < 16 * 128; idx += 256) {  // (kpair, n)
                int k = (idx >> 7) * 2, n = idx & 127;
                float v0 = W[(long)(kk + k)     * HID + n];
                float v1 = W[(long)(kk + k + 1) * HID + n];
                int fl = ((k >> 4) << 4) + (n & 15);
                *(unsigned int*)&BsF[((n >> 4) * 32 + fl) * 16 + (k & 15)] =
                    pack2bf(v0, v1);
            }
        } else {
            // ---- K-tail path (at most once per GEMM): zero-pad past K ----
            #pragma unroll
            for (int idx = tid; idx < 128 * 16; idx += 256) {
                int r = idx >> 4, k = (idx & 15) * 2;
                int gr = rowB + r; gr = gr > rMax ? rMax : gr;
                long base = (long)gr * K + kk + k;
                float v0 = ((kk + k)     < K) ? A[base]     : 0.0f;
                float v1 = ((kk + k + 1) < K) ? A[base + 1] : 0.0f;
                int h = k >> 4;
                int e = k & 15;
                *(unsigned int*)&AsF[r * 40 + h * 16 + e] = pack2bf(v0, v1);
            }
            #pragma unroll
            for (int idx = tid; idx < 16 * 128; idx += 256) {
                int k = (idx >> 7) * 2, n = idx & 127;
                int gk = kk + k;
                float v0 = (gk     < K) ? W[(long)gk * HID + n]       : 0.0f;
                float v1 = (gk + 1 < K) ? W[(long)(gk + 1) * HID + n] : 0.0f;
                int fl = ((k >> 4) << 4) + (n & 15);
                *(unsigned int*)&BsF[((n >> 4) * 32 + fl) * 16 + (k & 15)] =
                    pack2bf(v0, v1);
            }
        }
        // speculative prefetch of next A K-chunk (global_prefetch_b8)
        if (kk + 32 < K && tid < 128) {
            int gr = rowB + tid; gr = gr > rMax ? rMax : gr;
            __builtin_prefetch(&A[(long)gr * K + kk + 32], 0, 0);
        }
        __syncthreads();

        // --- A fragment + all 8 B fragments: burst of ds_load_b128 into
        //     distinct registers, so the WMMA chain retires on partial waits ---
        const unsigned short* ap = &AsF[(wrow + l15) * 40 + half * 16];
        F16frag fa;
        fa.q[0] = *(const uint4*)ap;
        fa.q[1] = *(const uint4*)(ap + 8);

        F16frag fb[8];
        #pragma unroll
        for (int c = 0; c < 8; ++c) {
            const unsigned short* bp = &BsF[(c * 32 + lane) * 16];
            fb[c].q[0] = *(const uint4*)bp;
            fb[c].q[1] = *(const uint4*)(bp + 8);
        }

        v16bf a = __builtin_bit_cast(v16bf, fa.u);
        #pragma unroll
        for (int c = 0; c < 8; ++c) {
            v16bf b = __builtin_bit_cast(v16bf, fb[c].u);
            acc[c] = __builtin_amdgcn_wmma_f32_16x16x32_bf16(
                false, a, false, b, (short)0, acc[c], false, false);
        }
        __syncthreads();
    }

    // --- epilogue: bias (+C) (+ReLU); D layout: VGPR v -> row v+8*half ---
    if (rowB + 128 <= M) {
        #pragma unroll
        for (int c = 0; c < 8; ++c) {
            #pragma unroll
            for (int v = 0; v < 8; ++v) {
                int  row = rowB + wrow + v + 8 * half;
                int  col = c * 16 + l15;
                long off = (long)row * HID + col;
                float val = acc[c][v] + bias[col];
                if (ACC)  val += C[off];
                if (RELU) val = fmaxf(val, 0.0f);
                C[off] = val;
            }
        }
    } else {
        #pragma unroll
        for (int c = 0; c < 8; ++c) {
            #pragma unroll
            for (int v = 0; v < 8; ++v) {
                int row = rowB + wrow + v + 8 * half;
                int col = c * 16 + l15;
                if (row < M) {
                    long off = (long)row * HID + col;
                    float val = acc[c][v] + bias[col];
                    if (ACC)  val += C[off];
                    if (RELU) val = fmaxf(val, 0.0f);
                    C[off] = val;
                }
            }
        }
    }
}

// ---------------------------------------------------------------------------
// Aggregation helpers
// ---------------------------------------------------------------------------
__global__ void zero_kernel(float* __restrict__ p, long n) {
    long t = (long)blockIdx.x * blockDim.x + threadIdx.x;
    if (t < n) p[t] = 0.0f;
}

__global__ void scatter_add_kernel(const float* __restrict__ X,
                                   const int* __restrict__ src,
                                   const int* __restrict__ dst,
                                   float* __restrict__ agg, int E)
{
    long t = (long)blockIdx.x * blockDim.x + threadIdx.x;
    if (t >= (long)E * HID) return;
    int e = (int)(t >> 7);
    int h = (int)(t & 127);
    float v = X[(long)src[e] * HID + h];
    atomicAdd(&agg[(long)dst[e] * HID + h], v);
}

__global__ void count_kernel(const int* __restrict__ dst,
                             float* __restrict__ cnt, int E)
{
    int e = blockIdx.x * blockDim.x + threadIdx.x;
    if (e < E) atomicAdd(&cnt[dst[e]], 1.0f);
}

__global__ void mean_kernel(float* __restrict__ agg,
                            const float* __restrict__ cnt, int N)
{
    long t = (long)blockIdx.x * blockDim.x + threadIdx.x;
    if (t >= (long)N * HID) return;
    int i = (int)(t >> 7);
    agg[t] *= __frcp_rn(fmaxf(cnt[i], 1.0f));
}

// ---------------------------------------------------------------------------
// Host-side launchers
// ---------------------------------------------------------------------------
static inline void launch_gemm(const float* A, int K, const float* W,
                               const float* b, float* C, int M,
                               int acc, int relu, hipStream_t s)
{
    dim3 g((M + 127) / 128), blk(256);
    if (acc) {
        if (relu) gemm_bf16_wmma<1, 1><<<g, blk, 0, s>>>(A, K, W, b, C, M);
        else      gemm_bf16_wmma<1, 0><<<g, blk, 0, s>>>(A, K, W, b, C, M);
    } else {
        if (relu) gemm_bf16_wmma<0, 1><<<g, blk, 0, s>>>(A, K, W, b, C, M);
        else      gemm_bf16_wmma<0, 0><<<g, blk, 0, s>>>(A, K, W, b, C, M);
    }
}

static inline void aggregate_mean(const float* X, const int* src, const int* dst,
                                  int E, int Ndst, float* agg, float* cnt,
                                  hipStream_t s)
{
    long nAgg = (long)Ndst * HID;
    zero_kernel<<<dim3((unsigned)((nAgg + 255) / 256)), dim3(256), 0, s>>>(agg, nAgg);
    zero_kernel<<<dim3((Ndst + 255) / 256), dim3(256), 0, s>>>(cnt, (long)Ndst);
    long nSc = (long)E * HID;
    scatter_add_kernel<<<dim3((unsigned)((nSc + 255) / 256)), dim3(256), 0, s>>>(X, src, dst, agg, E);
    count_kernel<<<dim3((E + 255) / 256), dim3(256), 0, s>>>(dst, cnt, E);
    mean_kernel<<<dim3((unsigned)((nAgg + 255) / 256)), dim3(256), 0, s>>>(agg, cnt, Ndst);
}

extern "C" void kernel_launch(void* const* d_in, const int* in_sizes, int n_in,
                              void* d_out, int out_size, void* d_ws, size_t ws_size,
                              hipStream_t stream)
{
    // ---- inputs (setup_inputs dict order) ----
    const float* x_user  = (const float*)d_in[0];
    const float* x_post  = (const float*)d_in[1];
    const float* x_sub   = (const float*)d_in[2];
    const int* src_up = (const int*)d_in[3];  const int* dst_up = (const int*)d_in[4];
    const int* src_ru = (const int*)d_in[5];  const int* dst_ru = (const int*)d_in[6];
    const int* src_bt = (const int*)d_in[7];  const int* dst_bt = (const int*)d_in[8];
    const int* src_rb = (const int*)d_in[9];  const int* dst_rb = (const int*)d_in[10];
    const float* Wr_user = (const float*)d_in[11]; const float* br_user = (const float*)d_in[12];
    const float* Wr_post = (const float*)d_in[13]; const float* br_post = (const float*)d_in[14];
    const float* Wr_sub  = (const float*)d_in[15]; const float* br_sub  = (const float*)d_in[16];
    const float* W1_up = (const float*)d_in[17]; const float* b1_up = (const float*)d_in[18];
    const float* W1_ru = (const float*)d_in[19]; const float* b1_ru = (const float*)d_in[20];
    const float* W1_bt = (const float*)d_in[21]; const float* b1_bt = (const float*)d_in[22];
    const float* W1_rb = (const float*)d_in[23]; const float* b1_rb = (const float*)d_in[24];
    const float* W2_up = (const float*)d_in[25]; const float* b2_up = (const float*)d_in[26];
    const float* W2_ru = (const float*)d_in[27]; const float* b2_ru = (const float*)d_in[28];
    const float* W2_bt = (const float*)d_in[29]; const float* b2_bt = (const float*)d_in[30];
    const float* W2_rb = (const float*)d_in[31]; const float* b2_rb = (const float*)d_in[32];
    const float* Wl_user = (const float*)d_in[33]; const float* bl_user = (const float*)d_in[34];
    const float* Wl_post = (const float*)d_in[35]; const float* bl_post = (const float*)d_in[36];
    const float* Wl_sub  = (const float*)d_in[37]; const float* bl_sub  = (const float*)d_in[38];

    // ---- workspace layout (floats) ----
    float* ws  = (float*)d_ws;
    float* hAu = ws;                       // NU*128  layer-A hidden states
    float* hAp = hAu + (size_t)NU * HID;   // NP*128
    float* hAs = hAp + (size_t)NP * HID;   // NS*128
    float* hBu = hAs + (size_t)NS * HID;   // layer-B hidden states
    float* hBp = hBu + (size_t)NU * HID;
    float* hBs = hBp + (size_t)NP * HID;
    float* agg = hBs + (size_t)NS * HID;   // NP*128 scratch (largest dest)
    float* cnt = agg + (size_t)NP * HID;   // NP counts

    // ---- outputs: [out_user | out_post | out_sub] flat ----
    float* out_user = (float*)d_out;
    float* out_post = out_user + (size_t)NU * HID;
    float* out_sub  = out_post + (size_t)NP * HID;

    // ===== root transforms =====
    launch_gemm(x_user, 3,   Wr_user, br_user, hAu, NU, 0, 0, stream);
    launch_gemm(x_post, 770, Wr_post, br_post, hAp, NP, 0, 0, stream);
    launch_gemm(x_sub,  9,   Wr_sub,  br_sub,  hAs, NS, 0, 0, stream);

    // ===== conv layer 1 (A -> B), ReLU after per-type sum =====
    aggregate_mean(hAp, src_ru, dst_ru, E_UP, NU, agg, cnt, stream);
    launch_gemm(agg, HID, W1_ru, b1_ru, hBu, NU, 0, 1, stream);

    aggregate_mean(hAp, src_bt, dst_bt, E_BT, NS, agg, cnt, stream);
    launch_gemm(agg, HID, W1_bt, b1_bt, hBs, NS, 0, 1, stream);

    aggregate_mean(hAu, src_up, dst_up, E_UP, NP, agg, cnt, stream);
    launch_gemm(agg, HID, W1_up, b1_up, hBp, NP, 0, 0, stream);
    aggregate_mean(hAs, src_rb, dst_rb, E_BT, NP, agg, cnt, stream);
    launch_gemm(agg, HID, W1_rb, b1_rb, hBp, NP, 1, 1, stream);   // += , then ReLU

    // ===== conv layer 2 (B -> A), ReLU after per-type sum =====
    aggregate_mean(hBp, src_ru, dst_ru, E_UP, NU, agg, cnt, stream);
    launch_gemm(agg, HID, W2_ru, b2_ru, hAu, NU, 0, 1, stream);

    aggregate_mean(hBp, src_bt, dst_bt, E_BT, NS, agg, cnt, stream);
    launch_gemm(agg, HID, W2_bt, b2_bt, hAs, NS, 0, 1, stream);

    aggregate_mean(hBu, src_up, dst_up, E_UP, NP, agg, cnt, stream);
    launch_gemm(agg, HID, W2_up, b2_up, hAp, NP, 0, 0, stream);
    aggregate_mean(hBs, src_rb, dst_rb, E_BT, NP, agg, cnt, stream);
    launch_gemm(agg, HID, W2_rb, b2_rb, hAp, NP, 1, 1, stream);   // += , then ReLU

    // ===== final per-type linears =====
    launch_gemm(hAu, HID, Wl_user, bl_user, out_user, NU, 0, 0, stream);
    launch_gemm(hAp, HID, Wl_post, bl_post, out_post, NP, 0, 0, stream);
    launch_gemm(hAs, HID, Wl_sub,  bl_sub,  out_sub,  NS, 0, 0, stream);
}
